// GaussianMixtureModel_17171279249501
// MI455X (gfx1250) — matrix operand order repaired and checked by reading, not compiled, not run on hardware
//
#include <hip/hip_runtime.h>
#include <hip/hip_bf16.h>
#include <math.h>

// ---------------------------------------------------------------------------
// GMM E-step on MI455X (gfx1250, wave32, WMMA).
// Dominant cost: batched GEMM  y_k = X @ L_k  (K=64 GEMMs of 131072x256x256,
// 1.1 TFLOP total) -> compute-bound (HBM traffic only ~190MB ~ 8us).
// Strategy: bf16x3 emulated-fp32 GEMM on v_wmma_f32_16x16x32_bf16.
// Round 2: repartition work so each wave owns 2 column tiles x 8 row strips:
//   - B fragments hoisted to registers once per K-slice (8x reuse)
//   - per-wave ds_load_b128 drops 68 -> 40 per 48 WMMAs
//   - two independent accumulator chains (j0/j1) for XDL pipelining
// ---------------------------------------------------------------------------

#define GMM_N 131072
#define GMM_K 64
#define GMM_D 256

typedef __attribute__((ext_vector_type(16))) __bf16 v16bf;
typedef __attribute__((ext_vector_type(8)))  __bf16 v8bf;
typedef __attribute__((ext_vector_type(8)))  float  v8f;

#define WMMA_BF16(A, B, C) \
    __builtin_amdgcn_wmma_f32_16x16x32_bf16(false, (A), false, (B), (short)0, (C), false, false)

// ---------------------------------------------------------------------------
// Kernel 1: per-component constants.
//   mu_proj[k][d] = sum_i means[k][i] * L_k[i][d]
//   constk[k]     = sum_d log(L_k[d][d]) + log(w[k]) - 0.5*D*log(2*pi)
// ---------------------------------------------------------------------------
__global__ void gmm_setup(const float* __restrict__ weights,
                          const float* __restrict__ means,
                          const float* __restrict__ prec,
                          float* __restrict__ mu_proj,
                          float* __restrict__ constk) {
    const int k = blockIdx.x;
    const int d = threadIdx.x;                 // 0..255
    const float* Lk = prec + (size_t)k * GMM_D * GMM_D;
    const float* mu = means + k * GMM_D;

    float acc = 0.f;
    for (int i = 0; i < GMM_D; ++i)            // column-d dot product (coalesced over d)
        acc += mu[i] * Lk[(size_t)i * GMM_D + d];
    mu_proj[k * GMM_D + d] = acc;

    __shared__ float red[GMM_D];
    red[d] = logf(Lk[(size_t)d * GMM_D + d]);
    __syncthreads();
    for (int s = 128; s > 0; s >>= 1) {
        if (d < s) red[d] += red[d + s];
        __syncthreads();
    }
    if (d == 0) {
        // -0.5 * D * log(2*pi), D = 256
        constk[k] = red[0] + logf(weights[k]) - 0.5f * (float)GMM_D * 1.8378770664093453f;
    }
}

// ---------------------------------------------------------------------------
// Kernel 2: main WMMA GEMM + squared-norm epilogue.
// Grid: (K, N/128). Block: 256 threads = 8 waves. Wave w owns column tiles
// j = {2w, 2w+1} over all 8 row strips (128 rows).
// ---------------------------------------------------------------------------
__global__ __launch_bounds__(256)
void gmm_main(const float* __restrict__ X,
              const float* __restrict__ prec,
              const float* __restrict__ mu_proj,
              const float* __restrict__ constk,
              float* __restrict__ out_resp) {
    __shared__ __bf16 sXhi[128][32];   //  8 KB
    __shared__ __bf16 sXlo[128][32];   //  8 KB
    __shared__ __bf16 sBhi[256][32];   // 16 KB (L slice, transposed: [col][k'])
    __shared__ __bf16 sBlo[256][32];   // 16 KB
    __shared__ float  sPart[8][128];   //  4 KB  -> 52 KB total static LDS

    const int k       = blockIdx.x;
    const int rowBase = blockIdx.y * 128;
    const int tid     = threadIdx.x;
    const int lane    = tid & 31;
    const int wave    = tid >> 5;
    const int lane15  = lane & 15;
    const int half    = lane >> 4;
    const int j0      = wave << 1;     // this wave's two column tiles
    const int j1      = j0 + 1;

    const float* Lk = prec + (size_t)k * GMM_D * GMM_D;

    v8f acc[8][2];
    const v8f vzero = {};
    #pragma unroll
    for (int s = 0; s < 8; ++s) { acc[s][0] = vzero; acc[s][1] = vzero; }

    for (int kk = 0; kk < GMM_D; kk += 32) {
        __syncthreads();   // previous iteration's fragment reads done before restage

        // ---- stage X slice: 128 rows x 32 cols, f32 -> bf16 hi/lo ----------
        #pragma unroll
        for (int it = 0; it < 4; ++it) {
            int idx = tid + it * 256;          // 1024 float4 total
            int r   = idx >> 3;                // 8 float4 per row
            int c4  = (idx & 7) << 2;
            const float4 v = *(const float4*)(X + (size_t)(rowBase + r) * GMM_D + kk + c4);
            float xs[4] = {v.x, v.y, v.z, v.w};
            #pragma unroll
            for (int u = 0; u < 4; ++u) {
                __bf16 h = (__bf16)xs[u];
                sXhi[r][c4 + u] = h;
                sXlo[r][c4 + u] = (__bf16)(xs[u] - (float)h);
            }
        }
        // ---- stage L slice rows kk..kk+31 x 256 cols, transposed ----------
        #pragma unroll
        for (int it = 0; it < 8; ++it) {
            int idx = tid + it * 256;          // 2048 float4 total
            int dp  = idx >> 6;                // 64 float4 per row -> 32 rows
            int c4  = (idx & 63) << 2;
            const float4 v = *(const float4*)(Lk + (size_t)(kk + dp) * GMM_D + c4);
            float xs[4] = {v.x, v.y, v.z, v.w};
            #pragma unroll
            for (int u = 0; u < 4; ++u) {
                __bf16 h = (__bf16)xs[u];
                sBhi[c4 + u][dp] = h;
                sBlo[c4 + u][dp] = (__bf16)(xs[u] - (float)h);
            }
        }
        __syncthreads();

        // ---- B fragments for this wave's 2 column tiles: hoisted, 8x reuse.
        // B k order used here (same as A): lane n=lane&15 holds contiguous K half (0..15/16..31).
        const int bko = half << 4;
        v16bf bh0 = *(const v16bf*)&sBhi[(j0 << 4) + lane15][bko];
        v16bf bl0 = *(const v16bf*)&sBlo[(j0 << 4) + lane15][bko];
        v16bf bh1 = *(const v16bf*)&sBhi[(j1 << 4) + lane15][bko];
        v16bf bl1 = *(const v16bf*)&sBlo[(j1 << 4) + lane15][bko];

        // ---- A layout: lanes<16 K{0..15}, lanes>=16 K{16..31} (matches B)
        const int koff = half ? 16 : 0;
        #pragma unroll
        for (int s = 0; s < 8; ++s) {
            const int arow = (s << 4) + lane15;
            v8bf a0h = *(const v8bf*)&sXhi[arow][koff];
            v8bf a1h = *(const v8bf*)&sXhi[arow][koff + 8];
            v8bf a0l = *(const v8bf*)&sXlo[arow][koff];
            v8bf a1l = *(const v8bf*)&sXlo[arow][koff + 8];
            v16bf ah = __builtin_shufflevector(a0h, a1h,
                        0,1,2,3,4,5,6,7,8,9,10,11,12,13,14,15);
            v16bf al = __builtin_shufflevector(a0l, a1l,
                        0,1,2,3,4,5,6,7,8,9,10,11,12,13,14,15);
            // bf16x3: hi*hi + hi*lo + lo*hi; interleave j0/j1 chains for ILP
            acc[s][0] = WMMA_BF16(ah, bh0, acc[s][0]);
            acc[s][1] = WMMA_BF16(ah, bh1, acc[s][1]);
            acc[s][0] = WMMA_BF16(ah, bl0, acc[s][0]);
            acc[s][1] = WMMA_BF16(ah, bl1, acc[s][1]);
            acc[s][0] = WMMA_BF16(al, bh0, acc[s][0]);
            acc[s][1] = WMMA_BF16(al, bh1, acc[s][1]);
        }
    }

    // ---- epilogue: per-wave partial column sums of (y - muL)^2 -------------
    // C/D layout: VGPR r -> row M = s*16 + 8*half + r, col N = lane&15.
    const float m0 = mu_proj[k * GMM_D + (j0 << 4) + lane15];
    const float m1 = mu_proj[k * GMM_D + (j1 << 4) + lane15];
    #pragma unroll
    for (int s = 0; s < 8; ++s) {
        #pragma unroll
        for (int r = 0; r < 8; ++r) {
            float t0 = acc[s][0][r] - m0;
            float t1 = acc[s][1][r] - m1;
            float v  = t0 * t0 + t1 * t1;
            #pragma unroll
            for (int mask = 1; mask <= 8; mask <<= 1)   // reduce 16-lane group
                v += __shfl_xor(v, mask, 32);
            if (lane15 == 0)
                sPart[wave][(s << 4) + (half << 3) + r] = v;
        }
    }
    __syncthreads();

    // cross-wave reduction: thread t (0..127) owns row t
    if (tid < 128) {
        float sum = 0.f;
        #pragma unroll
        for (int w2 = 0; w2 < 8; ++w2) sum += sPart[w2][tid];
        out_resp[(size_t)(rowBase + tid) * GMM_K + k] = constk[k] - 0.5f * sum;
    }
}

// ---------------------------------------------------------------------------
// Kernel 3: per-row logsumexp over K=64, in-place log_resp, block partial sums.
// One wave per row; 8 rows per block. Deterministic (no atomics).
// ---------------------------------------------------------------------------
__global__ void gmm_finalize(float* __restrict__ resp,
                             float* __restrict__ partials) {
    const int tid  = threadIdx.x;
    const int lane = tid & 31;
    const int wave = tid >> 5;
    const int n    = blockIdx.x * 8 + wave;
    float* row = resp + (size_t)n * GMM_K;

    float a = row[lane];
    float b = row[lane + 32];
    float m = fmaxf(a, b);
    #pragma unroll
    for (int mask = 1; mask <= 16; mask <<= 1)
        m = fmaxf(m, __shfl_xor(m, mask, 32));
    float s = expf(a - m) + expf(b - m);
    #pragma unroll
    for (int mask = 1; mask <= 16; mask <<= 1)
        s += __shfl_xor(s, mask, 32);
    const float lse = m + logf(s);

    row[lane]      = a - lse;
    row[lane + 32] = b - lse;

    __shared__ float sp[8];
    if (lane == 0) sp[wave] = lse;
    __syncthreads();
    if (tid == 0) {
        float t = 0.f;
        #pragma unroll
        for (int i = 0; i < 8; ++i) t += sp[i];
        partials[blockIdx.x] = t;
    }
}

// ---------------------------------------------------------------------------
// Kernel 4: deterministic reduction of 16384 partials -> mean in d_out[0].
// ---------------------------------------------------------------------------
__global__ void gmm_reduce(const float* __restrict__ partials,
                           float* __restrict__ out0) {
    __shared__ float red[256];
    float s = 0.f;
    for (int i = threadIdx.x; i < GMM_N / 8; i += 256) s += partials[i];
    red[threadIdx.x] = s;
    __syncthreads();
    for (int t = 128; t > 0; t >>= 1) {
        if (threadIdx.x < t) red[threadIdx.x] += red[threadIdx.x + t];
        __syncthreads();
    }
    if (threadIdx.x == 0) out0[0] = red[0] / (float)GMM_N;
}

// ---------------------------------------------------------------------------
extern "C" void kernel_launch(void* const* d_in, const int* in_sizes, int n_in,
                              void* d_out, int out_size, void* d_ws, size_t ws_size,
                              hipStream_t stream) {
    const float* X     = (const float*)d_in[0];   // (N, D)
    const float* w     = (const float*)d_in[1];   // (K,)
    const float* means = (const float*)d_in[2];   // (K, D)
    const float* prec  = (const float*)d_in[3];   // (K, D, D)
    float* out = (float*)d_out;                   // [0]=mean, [1..]=log_resp (N,K)

    float* mu_proj  = (float*)d_ws;               // K*D floats
    float* constk   = mu_proj + GMM_K * GMM_D;    // K floats
    float* partials = constk + GMM_K;             // N/8 floats

    gmm_setup<<<GMM_K, GMM_D, 0, stream>>>(w, means, prec, mu_proj, constk);
    gmm_main<<<dim3(GMM_K, GMM_N / 128), 256, 0, stream>>>(X, prec, mu_proj, constk, out + 1);
    gmm_finalize<<<GMM_N / 8, 256, 0, stream>>>(out + 1, partials);
    gmm_reduce<<<1, 256, 0, stream>>>(partials, out);
}
